// Head_8504035246173
// MI455X (gfx1250) — matrix operand order repaired and checked by reading, not compiled, not run on hardware
//
#include <hip/hip_runtime.h>

// ---------------------------------------------------------------------------
// MI455X / gfx1250 implementation.
// Heavy GEMMs (mafg c3 1x1-conv and the per-class linear einsum) run on
// v_wmma_f32_16x16x32_bf16 with the packed B operand held stationary in LDS
// (128 KB/block out of the 320 KB CDNA5 WGP LDS).  Everything else is cheap
// scalar fp32.
// ---------------------------------------------------------------------------

typedef __bf16 bf16_t;
typedef __attribute__((ext_vector_type(16))) __bf16 v16bf;
typedef __attribute__((ext_vector_type(8)))  float  v8f;
typedef unsigned int u32x4 __attribute__((ext_vector_type(4)));

#define B_    64
#define C_    512
#define H_    14
#define W_    14
#define HW_   196
#define PPAD_ 208     // 13 * 16 (padded pixel rows for WMMA M tiles)
#define NCLS_ 12
#define REL_  64
#define INVS_ 0.9999950000374997f   // 1/sqrt(1 + 1e-5)
#define LDS_BYTES_ 131072           // 8 N-tiles x 16 K-tiles x 1KB

union BF16Pack { u32x4 u[2]; v16bf v; };

static __device__ __forceinline__ v8f zero8() {
  v8f z = {0.f,0.f,0.f,0.f,0.f,0.f,0.f,0.f};
  return z;
}

// A tile 16x32 (MxK) bf16, row-major in memory with row stride C_ elements.
// ISA layout: lane<16: M=lane, K = {0..7, 16..23}; lane>=16: M=lane-16, K = {8..15, 24..31}.
static __device__ __forceinline__ v16bf load_a16x32(const bf16_t* __restrict__ tile, int lane) {
  int r = lane & 15;
  int h = (lane >> 4) << 3;
  const bf16_t* p = tile + (size_t)r * C_ + h;
  BF16Pack t;
  t.u[0] = *(const u32x4*)(p);        // cols h   .. h+7
  t.u[1] = *(const u32x4*)(p + 16);   // cols h+16.. h+23
  return t.v;
}

// B tile 32x16 (KxN) pre-packed as two 512B half-planes per tile:
//   halves 0..7 of each lane at tile*512 + lane*8, halves 8..15 at +256.
// Lane stride is 16B per ds_load_b128 -> conflict-free LDS reads.
static __device__ __forceinline__ v16bf load_b_lds(const bf16_t* __restrict__ Bs,
                                                   int tl, int lane) {
  const bf16_t* p = Bs + (size_t)tl * 512 + lane * 8;
  BF16Pack t;
  t.u[0] = *(const u32x4*)(p);
  t.u[1] = *(const u32x4*)(p + 256);
  return t.v;
}

static __device__ __forceinline__ v8f wmma_bf16(v16bf a, v16bf b, v8f c) {
  return __builtin_amdgcn_wmma_f32_16x16x32_bf16(false, a, false, b, (short)0, c, false, false);
}

// --------------------------- weight packing --------------------------------
// Pack W (O x C row-major f32) into WMMA B-operand bf16 tiles.
// Lane semantics per tile (ot,kt): n = lane&15, khalf = lane>>4,
// halves h = 0..7 -> W[ot*16 + n][kt*32 + khalf*8 + h], h = 8..15 -> k + 16 + khalf*8 + (h-8), stored as two
// 256-element half-planes (see load_b_lds).
__global__ __launch_bounds__(256) void k_pack(const float* __restrict__ Wall,
                                              bf16_t* __restrict__ Pall) {
  int mat = blockIdx.y;
  const float* Wm = Wall + (size_t)mat * C_ * C_;
  bf16_t*      Pm = Pall + (size_t)mat * C_ * C_;
  int t = blockIdx.x * blockDim.x + threadIdx.x;     // 16384 per matrix
  int lane = t & 31;
  int tile = t >> 5;                                  // ot * 16 + kt (ktiles = 16)
  int ot = tile >> 4, kt = tile & 15;
  int o  = ot * 16 + (lane & 15);
  int k0 = kt * 32 + ((lane >> 4) << 3);
  const float* src = Wm + (size_t)o * C_ + k0;
  bf16_t* dst0 = Pm + (size_t)tile * 512 + lane * 8;
  bf16_t* dst1 = dst0 + 256;
#pragma unroll
  for (int h = 0; h < 8; ++h) {
    dst0[h] = (bf16_t)src[h];
    dst1[h] = (bf16_t)src[16 + h];
  }
}

// --------------------------- MAFG stencil chain ----------------------------
// Per (b,c) plane: attn0 = 5x5 dwconv; + three separable (1xk)(kx1) branches.
// Writes s transposed to sT[b, p, c] as bf16 (A operand of GEMM #1).
__global__ __launch_bounds__(256) void k_stencil(
    const float* __restrict__ x,
    const float* __restrict__ c0w,  const float* __restrict__ c0b,
    const float* __restrict__ c01w, const float* __restrict__ c01b,
    const float* __restrict__ c02w, const float* __restrict__ c02b,
    const float* __restrict__ c11w, const float* __restrict__ c11b,
    const float* __restrict__ c12w, const float* __restrict__ c12b,
    const float* __restrict__ c21w, const float* __restrict__ c21b,
    const float* __restrict__ c22w, const float* __restrict__ c22b,
    bf16_t* __restrict__ sT)
{
  int bc = blockIdx.x;
  int b = bc >> 9, c = bc & (C_ - 1);
  __shared__ float xin[HW_], attn0[HW_], tmp[HW_];
  __shared__ float w0[25], wr[39], wc[39];
  int t = threadIdx.x;
  if (t < HW_) xin[t] = x[(size_t)bc * HW_ + t];
  if (t < 25)  w0[t]  = c0w[c * 25 + t];
  if (t >= 64  && t < 71)  { wr[t-64]      = c01w[c*7  + (t-64)];  wc[t-64]      = c02w[c*7  + (t-64)];  }
  if (t >= 96  && t < 107) { wr[7 + t-96]  = c11w[c*11 + (t-96)];  wc[7 + t-96]  = c12w[c*11 + (t-96)];  }
  if (t >= 128 && t < 149) { wr[18 + t-128]= c21w[c*21 + (t-128)]; wc[18 + t-128]= c22w[c*21 + (t-128)]; }
  __syncthreads();

  int h = t / W_, w = t % W_;
  float accv = 0.f;
  if (t < HW_) {
    float s = c0b[c];
#pragma unroll
    for (int kh = 0; kh < 5; ++kh) {
      int ih = h + kh - 2;
      if (ih < 0 || ih >= H_) continue;
#pragma unroll
      for (int kw = 0; kw < 5; ++kw) {
        int iw = w + kw - 2;
        if (iw >= 0 && iw < W_) s += w0[kh*5 + kw] * xin[ih*W_ + iw];
      }
    }
    attn0[t] = s;
    accv = s;
  }
  __syncthreads();

  const int klen[3] = {7, 11, 21};
  const int koff[3] = {0, 7, 18};
  for (int br = 0; br < 3; ++br) {
    int K = klen[br], base = koff[br], pad = K >> 1;
    float rb = (br == 0) ? c01b[c] : (br == 1) ? c11b[c] : c21b[c];
    float cb = (br == 0) ? c02b[c] : (br == 1) ? c12b[c] : c22b[c];
    if (t < HW_) {
      float s = rb;
      for (int k = 0; k < K; ++k) {
        int iw = w + k - pad;
        if (iw >= 0 && iw < W_) s += wr[base + k] * attn0[h*W_ + iw];
      }
      tmp[t] = s;
    }
    __syncthreads();
    if (t < HW_) {
      float s = cb;
      for (int k = 0; k < K; ++k) {
        int ih = h + k - pad;
        if (ih >= 0 && ih < H_) s += wc[base + k] * tmp[ih*W_ + w];
      }
      accv += s;
    }
    __syncthreads();
  }
  if (t < HW_) sT[((size_t)b * PPAD_ + t) * C_ + c] = (bf16_t)accv;
}

// zero padded rows 196..207 of sT
__global__ void k_pad(bf16_t* __restrict__ sT) {
  int i = blockIdx.x * blockDim.x + threadIdx.x;
  if (i >= B_ * 12 * C_) return;
  int c = i % C_, r = (i / C_) % 12, b = i / (C_ * 12);
  sT[((size_t)b * PPAD_ + HW_ + r) * C_ + c] = (bf16_t)0.f;
}

// --------------------------- GEMM #1 (WMMA, B stationary in LDS) -----------
// Y[p,o] = sum_c sT[b,p,c] * W3[o,c] + b3[o];  xs[b,p,o] = Y * x[b,o,p]  (bf16)
// grid: (b, ochunk=0..3); block = 4 waves; each block stages 8 N-tiles x all K
// of packed W3 (128KB) into LDS once; waves split the 13 M-tiles.
__global__ __launch_bounds__(128) void k_g1(
    const bf16_t* __restrict__ sT, const bf16_t* __restrict__ Wp,
    const float* __restrict__ b3, const float* __restrict__ x,
    bf16_t* __restrict__ xs)
{
  extern __shared__ char smem[];
  bf16_t* Bs = (bf16_t*)smem;
  int b  = blockIdx.x;
  int oc = blockIdx.y;                  // otiles oc*8 .. oc*8+7
  int t  = threadIdx.x;
  {
    const u32x4* src = (const u32x4*)(Wp + (size_t)oc * 8 * 16 * 512);
    u32x4* dst = (u32x4*)smem;
    for (int i = t; i < 8192; i += 128) dst[i] = src[i];   // 128KB contiguous
  }
  __syncthreads();
  int wv = t >> 5, lane = t & 31;

  for (int pass = 0; pass < 4; ++pass) {
    int pt = pass * 4 + wv;
    if (pt >= 13) continue;             // wave-uniform
    const bf16_t* A0 = sT + (size_t)b * PPAD_ * C_ + (size_t)pt * 16 * C_;
    v8f acc[8];
#pragma unroll
    for (int j = 0; j < 8; ++j) acc[j] = zero8();
    for (int k = 0; k < 16; ++k) {
      v16bf a = load_a16x32(A0 + k * 32, lane);
#pragma unroll
      for (int j = 0; j < 8; ++j) {
        v16bf bm = load_b_lds(Bs, j * 16 + k, lane);
        acc[j] = wmma_bf16(a, bm, acc[j]);
      }
    }
    int rbase = pt * 16 + ((lane >> 4) << 3);
#pragma unroll
    for (int j = 0; j < 8; ++j) {
      int o = (oc * 8 + j) * 16 + (lane & 15);
      float bias = b3[o];
      const float* xp = x + ((size_t)b * C_ + o) * HW_;
      bf16_t* op = xs + (size_t)b * PPAD_ * C_ + o;
#pragma unroll
      for (int e = 0; e < 8; ++e) {
        int p = rbase + e;
        float xvv = xp[p < HW_ ? p : 0];
        float v = (p < HW_) ? (acc[j][e] + bias) * xvv : 0.f;
        op[(size_t)p * C_] = (bf16_t)v;
      }
    }
  }
}

// --------------------------- GEMM #2 (WMMA + fused mean, B in LDS) ---------
// v_mafg[b,n,o] = mean_p relu(g*(xs[b] @ Wn^T + b)[p,o]*INVS + be)
// grid: (b*12+n, ochunk); waves split M-tiles; column sums reduced across
// waves through LDS (reusing the B buffer).
__global__ __launch_bounds__(128) void k_g2(
    const bf16_t* __restrict__ xs, const bf16_t* __restrict__ WpL,
    const float* __restrict__ lb, const float* __restrict__ lg,
    const float* __restrict__ lbe, float* __restrict__ vm)
{
  extern __shared__ char smem[];
  bf16_t* Bs = (bf16_t*)smem;
  int bn = blockIdx.x;                  // b * 12 + n
  int oc = blockIdx.y;                  // otiles oc*8 .. oc*8+7
  int b = bn / NCLS_, n = bn % NCLS_;
  int t = threadIdx.x;
  {
    const u32x4* src = (const u32x4*)(WpL + (size_t)n * C_ * C_ + (size_t)oc * 8 * 16 * 512);
    u32x4* dst = (u32x4*)smem;
    for (int i = t; i < 8192; i += 128) dst[i] = src[i];   // 128KB contiguous
  }
  __syncthreads();
  int wv = t >> 5, lane = t & 31;

  float colsum[8];
#pragma unroll
  for (int j = 0; j < 8; ++j) colsum[j] = 0.f;

  for (int pass = 0; pass < 4; ++pass) {
    int pt = pass * 4 + wv;
    if (pt >= 13) continue;             // wave-uniform
    const bf16_t* A0 = xs + (size_t)b * PPAD_ * C_ + (size_t)pt * 16 * C_;
    v8f acc[8];
#pragma unroll
    for (int j = 0; j < 8; ++j) acc[j] = zero8();
    for (int k = 0; k < 16; ++k) {
      v16bf a = load_a16x32(A0 + k * 32, lane);
#pragma unroll
      for (int j = 0; j < 8; ++j) {
        v16bf bm = load_b_lds(Bs, j * 16 + k, lane);
        acc[j] = wmma_bf16(a, bm, acc[j]);
      }
    }
    int rbase = pt * 16 + ((lane >> 4) << 3);
#pragma unroll
    for (int j = 0; j < 8; ++j) {
      int o = (oc * 8 + j) * 16 + (lane & 15);
      float bias = lb[n * C_ + o], gg = lg[n * C_ + o], bb = lbe[n * C_ + o];
#pragma unroll
      for (int e = 0; e < 8; ++e) {
        int p = rbase + e;
        if (p < HW_) {
          float y = acc[j][e] + bias;
          colsum[j] += fmaxf(gg * y * INVS_ + bb, 0.f);
        }
      }
    }
  }

  // cross-wave column-sum reduction (reuse LDS; B no longer needed)
  __syncthreads();
  float* red = (float*)smem;            // [4 waves][8 j][16 cols]
#pragma unroll
  for (int j = 0; j < 8; ++j) {
    float v = colsum[j] + __shfl_xor(colsum[j], 16, 32);   // lane L + lane L+16 share o
    if (lane < 16) red[(wv * 8 + j) * 16 + lane] = v;
  }
  __syncthreads();
  {
    int j = t >> 4, l15 = t & 15;       // t < 128 covers 8 x 16
    float s = red[(0 * 8 + j) * 16 + l15] + red[(1 * 8 + j) * 16 + l15] +
              red[(2 * 8 + j) * 16 + l15] + red[(3 * 8 + j) * 16 + l15];
    vm[(size_t)bn * C_ + (oc * 8 + j) * 16 + l15] = s * (1.f / HW_);
  }
}

// --------------------------- GNN -------------------------------------------
__global__ __launch_bounds__(192) void k_gnn_adj(const float* __restrict__ vm,
                                                 float* __restrict__ Agnn) {
  int b = blockIdx.x;
  __shared__ float V[NCLS_ * C_];
  __shared__ float si[NCLS_ * NCLS_], thrs[NCLS_], dinv[NCLS_];
  int t = threadIdx.x;
  for (int i = t; i < NCLS_ * C_; i += 192) V[i] = vm[(size_t)b * NCLS_ * C_ + i];
  __syncthreads();
  if (t < NCLS_ * NCLS_) {
    int i = t / NCLS_, j = t % NCLS_;
    float s = 0.f;
    for (int c = 0; c < C_; ++c) s += V[i * C_ + c] * V[j * C_ + c];
    si[t] = s;
  }
  __syncthreads();
  if (t < NCLS_) {
    float row[NCLS_];
    for (int j = 0; j < NCLS_; ++j) row[j] = si[t * NCLS_ + j];
    float thr = -1e30f;
    for (int s = 0; s < 4; ++s) {               // 4th largest (topk threshold)
      int bi = 0; float bv = -1e30f;
      for (int j = 0; j < NCLS_; ++j) if (row[j] > bv) { bv = row[j]; bi = j; }
      thr = bv; row[bi] = -1e30f;
    }
    thrs[t] = thr;
    float d = 0.f;
    for (int j = 0; j < NCLS_; ++j) d += (si[t * NCLS_ + j] >= thr) ? 1.f : 0.f;
    dinv[t] = rsqrtf(d);
  }
  __syncthreads();
  if (t < NCLS_ * NCLS_) {
    int i = t / NCLS_, j = t % NCLS_;
    float a = (si[t] >= thrs[i]) ? 1.f : 0.f;
    Agnn[(size_t)b * 144 + t] = dinv[i] * a * dinv[j];
  }
}

__global__ void k_gnn_vu(const float* __restrict__ vm,
                         const float* __restrict__ vW, const float* __restrict__ vb,
                         const float* __restrict__ uW, const float* __restrict__ ub,
                         float* __restrict__ Vx, float* __restrict__ Ux) {
  int i = blockIdx.x * blockDim.x + threadIdx.x;
  if (i >= B_ * NCLS_ * C_) return;
  int o = i % C_; size_t bn = (size_t)(i / C_);
  const float* vr = vm + bn * C_;
  float sv = vb[o], su = ub[o];
  for (int c = 0; c < C_; ++c) {
    float xv = vr[c];
    sv += vW[(size_t)o * C_ + c] * xv;
    su += uW[(size_t)o * C_ + c] * xv;
  }
  Vx[i] = sv; Ux[i] = su;
}

__global__ void k_gnn_agg(const float* __restrict__ vm, const float* __restrict__ Agnn,
                          const float* __restrict__ Vx, const float* __restrict__ Ux,
                          const float* __restrict__ g, const float* __restrict__ be,
                          float* __restrict__ vg) {
  int i = blockIdx.x * blockDim.x + threadIdx.x;
  if (i >= B_ * NCLS_ * C_) return;
  int o = i % C_; int n = (i / C_) % NCLS_; int b = i / (C_ * NCLS_);
  float s = Ux[i];
  const float* Ar  = Agnn + (size_t)b * 144 + n * NCLS_;
  const float* Vb0 = Vx + (size_t)b * NCLS_ * C_ + o;
  for (int j = 0; j < NCLS_; ++j) s += Ar[j] * Vb0[(size_t)j * C_];
  s = g[n] * s * INVS_ + be[n];
  vg[i] = fmaxf(vm[i] + s, 0.f);
}

// --------------------------- CCFG (factorized) -----------------------------
__global__ void k_xhsum(const float* __restrict__ x, float* __restrict__ xhsum) {
  int i = blockIdx.x * blockDim.x + threadIdx.x;      // (b,c,w)
  if (i >= B_ * C_ * W_) return;
  int w = i % W_; int bc = i / W_;
  const float* xp = x + (size_t)bc * HW_ + w;
  float s = 0.f;
  for (int h = 0; h < H_; ++h) s += xp[h * W_];
  xhsum[i] = s;
}

__global__ void k_x12(const float* __restrict__ xhsum,
                      const float* __restrict__ w1, const float* __restrict__ b1,
                      const float* __restrict__ w2, const float* __restrict__ b2,
                      float* __restrict__ x1, float* __restrict__ x2) {
  int i = blockIdx.x * blockDim.x + threadIdx.x;      // (b,r,u)
  if (i >= B_ * REL_ * W_) return;
  int u = i % W_; int r = (i / W_) % REL_; int b = i / (W_ * REL_);
  const float* xh = xhsum + (size_t)b * C_ * W_ + u;
  float a1 = 0.f, a2 = 0.f;
  for (int c = 0; c < C_; ++c) {
    float xv = xh[(size_t)c * W_];
    a1 += w1[(size_t)r * C_ + c] * xv;
    a2 += w2[(size_t)r * C_ + c] * xv;
  }
  x1[i] = a1 * (1.f / H_) + b1[r];
  x2[i] = a2 * (1.f / H_) + b2[r];
}

__global__ void k_tt(const float* __restrict__ x1, const float* __restrict__ x2,
                     float* __restrict__ Tt) {
  int i = blockIdx.x * blockDim.x + threadIdx.x;      // (b,r,v)
  if (i >= B_ * REL_ * W_) return;
  int v = i % W_;
  const float* x1r = x1 + (i - v);
  float xv = x2[i];
  float s = 0.f;
  for (int u = 0; u < W_; ++u) s += tanhf(x1r[u] - xv);
  Tt[i] = s;
}

__global__ void k_asum(const float* __restrict__ Tt, const float* __restrict__ w4,
                       const float* __restrict__ b4, float* __restrict__ Asum) {
  int i = blockIdx.x * blockDim.x + threadIdx.x;      // (b,o,v)
  if (i >= B_ * C_ * W_) return;
  int v = i % W_; int o = (i / W_) % C_; int b = i / (W_ * C_);
  const float* tr = Tt + (size_t)b * REL_ * W_ + v;
  float s = (float)W_ * b4[o];
  for (int r = 0; r < REL_; ++r) s += w4[(size_t)o * REL_ + r] * tr[(size_t)r * W_];
  Asum[i] = s;
}

__global__ void k_x3sum(const float* __restrict__ xhsum, const float* __restrict__ w3,
                        const float* __restrict__ b3c, float* __restrict__ X3sum) {
  int i = blockIdx.x * blockDim.x + threadIdx.x;      // (b,o,v)
  if (i >= B_ * C_ * W_) return;
  int v = i % W_; int o = (i / W_) % C_; int b = i / (W_ * C_);
  const float* xh = xhsum + (size_t)b * C_ * W_ + v;
  float s = (float)H_ * b3c[o];
  for (int c = 0; c < C_; ++c) s += w3[(size_t)o * C_ + c] * xh[(size_t)c * W_];
  X3sum[i] = s;
}

__global__ void k_pool(const float* __restrict__ Asum, const float* __restrict__ X3sum,
                       float* __restrict__ pool) {
  int i = blockIdx.x * blockDim.x + threadIdx.x;      // (b,c)
  if (i >= B_ * C_) return;
  float s = 0.f;
  for (int v = 0; v < W_; ++v) s += Asum[(size_t)i * W_ + v] * X3sum[(size_t)i * W_ + v];
  pool[i] = s * (1.f / HW_);
}

__global__ void k_clin(const float* __restrict__ pool, const float* __restrict__ Wm,
                       const float* __restrict__ bb, const float* __restrict__ g,
                       const float* __restrict__ be, float* __restrict__ outp) {
  int i = blockIdx.x * blockDim.x + threadIdx.x;      // (b,o)
  if (i >= B_ * C_) return;
  int o = i % C_; int b = i / C_;
  const float* pr = pool + (size_t)b * C_;
  float s = bb[o];
  for (int c = 0; c < C_; ++c) s += Wm[(size_t)o * C_ + c] * pr[c];
  outp[i] = fmaxf(g[o] * s * INVS_ + be[o], 0.f);
}

// --------------------------- fusion head -----------------------------------
__global__ void k_fuse(const float* __restrict__ vg, const float* __restrict__ cl,
                       const float* __restrict__ pcW, const float* __restrict__ pcb,
                       const float* __restrict__ pg, const float* __restrict__ pbe,
                       float* __restrict__ f) {
  int i = blockIdx.x * blockDim.x + threadIdx.x;      // (b,n,o)
  if (i >= B_ * NCLS_ * C_) return;
  int o = i % C_; size_t bn = (size_t)(i / C_); int b = (int)(bn / NCLS_);
  const float* g1 = vg + bn * C_;
  const float* g2 = cl + (size_t)b * C_;
  const float* wr = pcW + (size_t)o * 2 * C_;
  float s = pcb[o];
  for (int c = 0; c < C_; ++c) s += wr[c] * g1[c];
  for (int c = 0; c < C_; ++c) s += wr[C_ + c] * g2[c];
  f[i] = fmaxf(pg[o] * s * INVS_ + pbe[o], 0.f);
}

__global__ __launch_bounds__(256) void k_scn(const float* __restrict__ sc,
                                             float* __restrict__ scn) {
  int n = blockIdx.x, t = threadIdx.x;
  __shared__ float red[256];
  float ss = 0.f;
  for (int c = t; c < C_; c += 256) { float r = fmaxf(sc[n * C_ + c], 0.f); ss += r * r; }
  red[t] = ss; __syncthreads();
  for (int s = 128; s > 0; s >>= 1) { if (t < s) red[t] += red[t + s]; __syncthreads(); }
  float inv = 1.f / fmaxf(sqrtf(red[0]), 1e-12f);
  for (int c = t; c < C_; c += 256) scn[n * C_ + c] = fmaxf(sc[n * C_ + c], 0.f) * inv;
}

__global__ __launch_bounds__(256) void k_head(const float* __restrict__ f,
                                              const float* __restrict__ scn,
                                              float* __restrict__ out) {
  int bn = blockIdx.x; int n = bn % NCLS_;
  int t = threadIdx.x;
  __shared__ float r1[256], r2[256];
  const float* fr = f + (size_t)bn * C_;
  const float* sr = scn + (size_t)n * C_;
  float ss = 0.f, dd = 0.f;
  for (int c = t; c < C_; c += 256) { float v = fr[c]; ss += v * v; dd += v * sr[c]; }
  r1[t] = ss; r2[t] = dd; __syncthreads();
  for (int s = 128; s > 0; s >>= 1) {
    if (t < s) { r1[t] += r1[t + s]; r2[t] += r2[t + s]; }
    __syncthreads();
  }
  if (t == 0) out[bn] = r2[0] / fmaxf(sqrtf(r1[0]), 1e-12f);
}

// --------------------------- launch ----------------------------------------
extern "C" void kernel_launch(void* const* d_in, const int* in_sizes, int n_in,
                              void* d_out, int out_size, void* d_ws, size_t ws_size,
                              hipStream_t stream) {
  (void)in_sizes; (void)n_in; (void)out_size; (void)ws_size;
  const float* x    = (const float*)d_in[0];
  const float* c0w  = (const float*)d_in[1];  const float* c0b  = (const float*)d_in[2];
  const float* c01w = (const float*)d_in[3];  const float* c01b = (const float*)d_in[4];
  const float* c02w = (const float*)d_in[5];  const float* c02b = (const float*)d_in[6];
  const float* c11w = (const float*)d_in[7];  const float* c11b = (const float*)d_in[8];
  const float* c12w = (const float*)d_in[9];  const float* c12b = (const float*)d_in[10];
  const float* c21w = (const float*)d_in[11]; const float* c21b = (const float*)d_in[12];
  const float* c22w = (const float*)d_in[13]; const float* c22b = (const float*)d_in[14];
  const float* c3w  = (const float*)d_in[15]; const float* c3b  = (const float*)d_in[16];
  const float* linW = (const float*)d_in[17]; const float* linb = (const float*)d_in[18];
  const float* ling = (const float*)d_in[19]; const float* linbe= (const float*)d_in[20];
  const float* uW   = (const float*)d_in[21]; const float* ub   = (const float*)d_in[22];
  const float* vW   = (const float*)d_in[23]; const float* vb   = (const float*)d_in[24];
  const float* bng  = (const float*)d_in[25]; const float* bnbe = (const float*)d_in[26];
  const float* w1   = (const float*)d_in[27]; const float* b1   = (const float*)d_in[28];
  const float* w2   = (const float*)d_in[29]; const float* b2   = (const float*)d_in[30];
  const float* w3   = (const float*)d_in[31]; const float* b3c  = (const float*)d_in[32];
  const float* w4   = (const float*)d_in[33]; const float* b4   = (const float*)d_in[34];
  const float* clW  = (const float*)d_in[35]; const float* clb  = (const float*)d_in[36];
  const float* clg  = (const float*)d_in[37]; const float* clbe = (const float*)d_in[38];
  const float* pcW  = (const float*)d_in[39]; const float* pcb  = (const float*)d_in[40];
  const float* pcg  = (const float*)d_in[41]; const float* pcbe = (const float*)d_in[42];
  const float* sc   = (const float*)d_in[43];

  char* ws = (char*)d_ws;
  size_t off = 0;
  auto alloc = [&](size_t bytes) -> void* {
    void* p = ws + off;
    off = (off + bytes + 255) & ~(size_t)255;
    return p;
  };

  bf16_t* sT    = (bf16_t*)alloc((size_t)B_ * PPAD_ * C_ * 2);
  bf16_t* xs    = (bf16_t*)alloc((size_t)B_ * PPAD_ * C_ * 2);
  bf16_t* Wp3   = (bf16_t*)alloc((size_t)C_ * C_ * 2);
  bf16_t* WpL   = (bf16_t*)alloc((size_t)NCLS_ * C_ * C_ * 2);
  float* vmafg  = (float*)alloc((size_t)B_ * NCLS_ * C_ * 4);
  float* Agnn   = (float*)alloc((size_t)B_ * 144 * 4);
  float* Vx     = (float*)alloc((size_t)B_ * NCLS_ * C_ * 4);
  float* Ux     = (float*)alloc((size_t)B_ * NCLS_ * C_ * 4);
  float* vgnn   = (float*)alloc((size_t)B_ * NCLS_ * C_ * 4);
  float* xhsum  = (float*)alloc((size_t)B_ * C_ * W_ * 4);
  float* x1b    = (float*)alloc((size_t)B_ * REL_ * W_ * 4);
  float* x2b    = (float*)alloc((size_t)B_ * REL_ * W_ * 4);
  float* Ttb    = (float*)alloc((size_t)B_ * REL_ * W_ * 4);
  float* Asum   = (float*)alloc((size_t)B_ * C_ * W_ * 4);
  float* X3sum  = (float*)alloc((size_t)B_ * C_ * W_ * 4);
  float* pool   = (float*)alloc((size_t)B_ * C_ * 4);
  float* clout  = (float*)alloc((size_t)B_ * C_ * 4);
  float* fvec   = (float*)alloc((size_t)B_ * NCLS_ * C_ * 4);
  float* scn    = (float*)alloc((size_t)NCLS_ * C_ * 4);

  // weight packing for WMMA B operands
  k_pack<<<dim3(64, 1),  256, 0, stream>>>(c3w, Wp3);
  k_pack<<<dim3(64, 12), 256, 0, stream>>>(linW, WpL);

  // MAFG
  k_stencil<<<B_ * C_, 256, 0, stream>>>(x, c0w, c0b, c01w, c01b, c02w, c02b,
                                         c11w, c11b, c12w, c12b, c21w, c21b,
                                         c22w, c22b, sT);
  k_pad<<<(B_ * 12 * C_ + 255) / 256, 256, 0, stream>>>(sT);
  k_g1<<<dim3(B_, 4), 128, LDS_BYTES_, stream>>>(sT, Wp3, c3b, x, xs);
  k_g2<<<dim3(B_ * NCLS_, 4), 128, LDS_BYTES_, stream>>>(xs, WpL, linb, ling, linbe, vmafg);

  // GNN
  k_gnn_adj<<<B_, 192, 0, stream>>>(vmafg, Agnn);
  k_gnn_vu<<<(B_ * NCLS_ * C_ + 255) / 256, 256, 0, stream>>>(vmafg, vW, vb, uW, ub, Vx, Ux);
  k_gnn_agg<<<(B_ * NCLS_ * C_ + 255) / 256, 256, 0, stream>>>(vmafg, Agnn, Vx, Ux, bng, bnbe, vgnn);

  // CCFG (factorized through column sums)
  k_xhsum<<<(B_ * C_ * W_ + 255) / 256, 256, 0, stream>>>(x, xhsum);
  k_x12<<<(B_ * REL_ * W_ + 255) / 256, 256, 0, stream>>>(xhsum, w1, b1, w2, b2, x1b, x2b);
  k_tt<<<(B_ * REL_ * W_ + 255) / 256, 256, 0, stream>>>(x1b, x2b, Ttb);
  k_asum<<<(B_ * C_ * W_ + 255) / 256, 256, 0, stream>>>(Ttb, w4, b4, Asum);
  k_x3sum<<<(B_ * C_ * W_ + 255) / 256, 256, 0, stream>>>(xhsum, w3, b3c, X3sum);
  k_pool<<<(B_ * C_ + 255) / 256, 256, 0, stream>>>(Asum, X3sum, pool);
  k_clin<<<(B_ * C_ + 255) / 256, 256, 0, stream>>>(pool, clW, clb, clg, clbe, clout);

  // fusion + head
  k_fuse<<<(B_ * NCLS_ * C_ + 255) / 256, 256, 0, stream>>>(vgnn, clout, pcW, pcb, pcg, pcbe, fvec);
  k_scn<<<NCLS_, 256, 0, stream>>>(sc, scn);
  k_head<<<B_ * NCLS_, 256, 0, stream>>>(fvec, scn, (float*)d_out);
}
